// EncoderDecoderWrapper_50929722196743
// MI455X (gfx1250) — compile-verified, matrix operand order checked by reading, not run on hardware
//
#include <hip/hip_runtime.h>
#include <cstdint>
#include <cstddef>

// ---------------------------------------------------------------------------
// CDNA5 / gfx1250 GRU encoder-decoder around v_wmma_f32_16x16x32_bf16.
//
// Structure: recurrent scans (336+336+96 steps) run as PERSISTENT kernels
// (64 co-resident WGs) with a device-scope sense barrier between steps,
// eliminating ~768 per-step launches whose latency would dominate the
// sub-microsecond per-step GEMM (256x512 @ 512x1536, L2-resident operands).
// ---------------------------------------------------------------------------

typedef __attribute__((ext_vector_type(16))) __bf16 v16bf;
typedef __attribute__((ext_vector_type(8)))  float  v8f;

#define N_BATCH 256
#define T_LX    336
#define F_FF    64
#define T_LY    96
#define HID     512
#define G3      1536              // 3*HID
#define MROWS   (T_LY * N_BATCH)  // 24576 rows for head GEMMs

union FragBF { v16bf v; float4 q[2]; };

__device__ __forceinline__ v8f zero8() {
  v8f z;
#pragma unroll
  for (int i = 0; i < 8; ++i) z[i] = 0.0f;
  return z;
}

// 16-bit A-fragment (16x32 tile, row-major source). ISA layout: lane l(0..15)
// row M=l holds K = 0..7 and 16..23; lane l+16 holds K = 8..15 and 24..31.
// => two contiguous 8-element runs -> two global_load_b128 per lane.
// B operand (W^T) mirrors this, so reading rows of row-major W gives B frags.
__device__ __forceinline__ v16bf load_frag(const __bf16* __restrict__ base,
                                           int row0, int ld, int k, int lane) {
  const int half = lane >> 4;
  const int r    = lane & 15;
  const __bf16* p = base + (size_t)(row0 + r) * (size_t)ld + (size_t)(k + half * 8);
  FragBF f;
  f.q[0] = *reinterpret_cast<const float4*>(p);
  f.q[1] = *reinterpret_cast<const float4*>(p + 16);
  return f.v;
}

__device__ __forceinline__ v8f wmma_bf16(v16bf a, v16bf b, v8f c) {
  return __builtin_amdgcn_wmma_f32_16x16x32_bf16(false, a, false, b,
                                                 (short)0, c, false, false);
}

__device__ __forceinline__ float sigmoidf_(float x) {
  return 1.0f / (1.0f + __expf(-x));
}

// Device-scope sense barrier for a persistent grid of co-resident WGs.
// Release fence (flush to device-coherent L2) before arrival; acquire fence
// (invalidate WGP-local lines, cache shared by the whole WG) after release.
__device__ __forceinline__ void grid_barrier(unsigned* cnt,
                                             volatile unsigned* gen,
                                             unsigned nblocks) {
  __threadfence();   // release: make this thread's stores device-visible
  __syncthreads();
  if (threadIdx.x == 0) {
    const unsigned g = *gen;
    if (atomicAdd(cnt, 1u) == nblocks - 1u) {
      *cnt = 0u;
      __threadfence();
      atomicAdd((unsigned*)gen, 1u);
    } else {
      while (*gen == g) __builtin_amdgcn_s_sleep(2);
    }
  }
  __syncthreads();
  __threadfence();   // acquire: drop stale lines before next step's reads
}

// ---------------------------------------------------------------------------
// Persistent GRU scan: T fused steps, device barrier between steps.
//   gates = [x_t @ Wih^T + bih], [h @ Whh^T + bhh]
//   r=sig(gi_r+gh_r); z=sig(gi_z+gh_z); n=tanh(gi_n + r*gh_n); h'=(1-z)n+z h
// Grid: (HID/64, N_BATCH/32) = 64 WGs, block 256 (8 waves), wave tile 16x16.
// seq_mode=1: outputs the full sequence hs_seq (T,N,H).
// seq_mode=0: ping-pong pp0/pp1 (final state in pp1 when T is even).
// ---------------------------------------------------------------------------
__global__ __launch_bounds__(256) void k_gru_scan(
    const __bf16* __restrict__ x_seq, int Fx, int T, int seq_mode,
    const __bf16* __restrict__ Wih,   // (G3 x Fx)  rows r|z|n
    const __bf16* __restrict__ Whh,   // (G3 x HID) rows r|z|n
    const float*  __restrict__ bih,
    const float*  __restrict__ bhh,
    const __bf16* __restrict__ h_init,
    __bf16*       __restrict__ hs_seq,
    __bf16*       __restrict__ pp0,
    __bf16*       __restrict__ pp1,
    unsigned* bar_cnt, volatile unsigned* bar_gen)
{
  const int lane  = threadIdx.x & 31;
  const int wave  = threadIdx.x >> 5;
  const int wj    = wave & 3;
  const int wb    = wave >> 2;
  const int jbase = blockIdx.x * 64 + wj * 16;
  const int bbase = blockIdx.y * 32 + wb * 16;
  const unsigned nblocks = gridDim.x * gridDim.y;
  const size_t NH = (size_t)N_BATCH * HID;

  const int  half = lane >> 4;
  const int  j    = jbase + (lane & 15);
  const float bir = bih[j],          bhr = bhh[j];
  const float biz = bih[HID + j],    bhz = bhh[HID + j];
  const float bin = bih[2*HID + j],  bhn = bhh[2*HID + j];

  for (int t = 0; t < T; ++t) {
    const __bf16* hp;
    __bf16* ho;
    if (seq_mode) {
      hp = t ? (hs_seq + (size_t)(t - 1) * NH) : h_init;
      ho = hs_seq + (size_t)t * NH;
    } else {
      hp = t ? ((t & 1) ? pp0 : pp1) : h_init;
      ho = (t & 1) ? pp1 : pp0;
    }
    const __bf16* x = x_seq + (size_t)t * N_BATCH * Fx;

    v8f ar = zero8(), az = zero8(), aghn = zero8(), agin = zero8();

    // gh = h @ Whh^T (K = HID)
    for (int k = 0; k < HID; k += 32) {
      v16bf a  = load_frag(hp, bbase, HID, k, lane);
      v16bf br = load_frag(Whh,                         jbase, HID, k, lane);
      v16bf bz = load_frag(Whh + (size_t)HID * HID,     jbase, HID, k, lane);
      v16bf bn = load_frag(Whh + (size_t)2 * HID * HID, jbase, HID, k, lane);
      ar   = wmma_bf16(a, br, ar);
      az   = wmma_bf16(a, bz, az);
      aghn = wmma_bf16(a, bn, aghn);
    }
    // gi = x_t @ Wih^T (K = Fx); r,z share accumulators, n kept separate
    for (int k = 0; k < Fx; k += 32) {
      v16bf a  = load_frag(x, bbase, Fx, k, lane);
      v16bf br = load_frag(Wih,                        jbase, Fx, k, lane);
      v16bf bz = load_frag(Wih + (size_t)HID * Fx,     jbase, Fx, k, lane);
      v16bf bn = load_frag(Wih + (size_t)2 * HID * Fx, jbase, Fx, k, lane);
      ar   = wmma_bf16(a, br, ar);
      az   = wmma_bf16(a, bz, az);
      agin = wmma_bf16(a, bn, agin);
    }

    // C/D layout: lane&15 = N (j); vgpr v = M low 3 bits; half = M bit 3.
#pragma unroll
    for (int v = 0; v < 8; ++v) {
      const int b = bbase + half * 8 + v;
      const float r  = sigmoidf_(ar[v] + bir + bhr);
      const float z  = sigmoidf_(az[v] + biz + bhz);
      const float nn = tanhf(agin[v] + bin + r * (aghn[v] + bhn));
      const float hpv = (float)hp[(size_t)b * HID + j];
      const float hn  = (1.0f - z) * nn + z * hpv;
      ho[(size_t)b * HID + j] = (__bf16)hn;
    }

    grid_barrier(bar_cnt, bar_gen, nblocks);
  }
}

// ---------------------------------------------------------------------------
// Generic head GEMM: out = act(A(MxK) @ B^T + bias), B row-major (>=64 rows).
// Grid: ((Nstore+63)/64, M/32), block 256, wave tile 16x16.
// ---------------------------------------------------------------------------
template <bool RELU, bool OUTBF>
__global__ __launch_bounds__(256) void k_gemm(
    const __bf16* __restrict__ A, int K,
    const __bf16* __restrict__ B,
    const float*  __restrict__ bias, int Nstore,
    void* __restrict__ outv, int ldC)
{
  const int lane  = threadIdx.x & 31;
  const int wave  = threadIdx.x >> 5;
  const int wn    = wave & 3;
  const int wm    = wave >> 2;
  const int nbase = blockIdx.x * 64 + wn * 16;
  const int mbase = blockIdx.y * 32 + wm * 16;

  v8f acc = zero8();
  for (int k = 0; k < K; k += 32) {
    v16bf a = load_frag(A, mbase, K, k, lane);
    v16bf b = load_frag(B, nbase, K, k, lane);
    acc = wmma_bf16(a, b, acc);
  }

  const int half = lane >> 4;
  const int n    = nbase + (lane & 15);
  if (n < Nstore) {
    const float bv = bias ? bias[n] : 0.0f;
#pragma unroll
    for (int v = 0; v < 8; ++v) {
      const int m = mbase + half * 8 + v;
      float c = acc[v] + bv;
      if (RELU) c = fmaxf(c, 0.0f);
      if (OUTBF) ((__bf16*)outv)[(size_t)m * ldC + n] = (__bf16)c;
      else       ((float*)outv)[(size_t)m * ldC + n]  = c;
    }
  }
}

// ---------------------------------------------------------------------------
// Data prep / epilogue kernels
// ---------------------------------------------------------------------------
__global__ void k_convert_pad(const float* __restrict__ src, __bf16* __restrict__ dst,
                              int sr, int sc, int dr, int dc) {
  const int total = dr * dc;
  for (int i = blockIdx.x * blockDim.x + threadIdx.x; i < total;
       i += gridDim.x * blockDim.x) {
    const int r = i / dc, c = i % dc;
    const float v = (r < sr && c < sc) ? src[(size_t)r * sc + c] : 0.0f;
    dst[i] = (__bf16)v;
  }
}

__global__ void k_zero_bf(__bf16* __restrict__ dst, int total) {
  for (int i = blockIdx.x * blockDim.x + threadIdx.x; i < total;
       i += gridDim.x * blockDim.x)
    dst[i] = (__bf16)0.0f;
}

__global__ void k_init_bar(unsigned* __restrict__ cnt, unsigned* __restrict__ gen) {
  if (threadIdx.x == 0 && blockIdx.x == 0) { *cnt = 0u; *gen = 0u; }
}

// x_f (N,LX,FF) f32 -> xa (LX,N,FF) bf16
__global__ void k_pack_x(const float* __restrict__ x_f, __bf16* __restrict__ xa) {
  const int total = T_LX * N_BATCH * F_FF;
  for (int i = blockIdx.x * blockDim.x + threadIdx.x; i < total;
       i += gridDim.x * blockDim.x) {
    const int t   = i / (N_BATCH * F_FF);
    const int rem = i % (N_BATCH * F_FF);
    const int n   = rem / F_FF;
    const int f   = rem % F_FF;
    xa[i] = (__bf16)x_f[((size_t)n * T_LX + t) * F_FF + f];
  }
}

// dec_in (LY,N,32) bf16: [0:16]=y_t shifted, [16:26]=y[0]/p shifted, pad 0
__global__ void k_build_decin(const float* __restrict__ y_t, const float* __restrict__ y,
                              const float* __restrict__ p, __bf16* __restrict__ dst) {
  const int total = T_LY * N_BATCH * 32;
  for (int i = blockIdx.x * blockDim.x + threadIdx.x; i < total;
       i += gridDim.x * blockDim.x) {
    const int t   = i / (N_BATCH * 32);
    const int rem = i % (N_BATCH * 32);
    const int n   = rem / 32;
    const int c   = rem % 32;
    float v = 0.0f;
    if (c < 16) {
      const int st = (t == 0) ? 0 : (t - 1);
      v = y_t[((size_t)n * T_LY + st) * 16 + c];
    } else if (c < 26) {
      const int cc = c - 16;
      v = (t == 0) ? y[((size_t)n * T_LY) * 10 + cc]
                   : p[((size_t)n * T_LY + (t - 1)) * 10 + cc];
    }
    dst[i] = (__bf16)v;
  }
}

// out (N,LY,10) = concat(temphr (LY*N,2), cls (LY*N,8)) transposed
__global__ void k_assemble(const float* __restrict__ th, const float* __restrict__ cl,
                           float* __restrict__ out) {
  const int total = N_BATCH * T_LY * 10;
  for (int i = blockIdx.x * blockDim.x + threadIdx.x; i < total;
       i += gridDim.x * blockDim.x) {
    const int n   = i / (T_LY * 10);
    const int rem = i % (T_LY * 10);
    const int t   = rem / 10;
    const int c   = rem % 10;
    out[i] = (c < 2) ? th[((size_t)t * N_BATCH + n) * 2 + c]
                     : cl[((size_t)t * N_BATCH + n) * 8 + (c - 2)];
  }
}

// ---------------------------------------------------------------------------
extern "C" void kernel_launch(void* const* d_in, const int* in_sizes, int n_in,
                              void* d_out, int out_size, void* d_ws, size_t ws_size,
                              hipStream_t stream) {
  (void)in_sizes; (void)n_in; (void)out_size; (void)ws_size;

  const float* x_f  = (const float*)d_in[0];
  const float* y_t  = (const float*)d_in[2];
  const float* y_   = (const float*)d_in[3];
  const float* p_   = (const float*)d_in[4];
  const float* Wih0 = (const float*)d_in[5];
  const float* Whh0 = (const float*)d_in[6];
  const float* bih0 = (const float*)d_in[7];
  const float* bhh0 = (const float*)d_in[8];
  const float* Wih1 = (const float*)d_in[9];
  const float* Whh1 = (const float*)d_in[10];
  const float* bih1 = (const float*)d_in[11];
  const float* bhh1 = (const float*)d_in[12];
  const float* dWih = (const float*)d_in[13];
  const float* dWhh = (const float*)d_in[14];
  const float* dbih = (const float*)d_in[15];
  const float* dbhh = (const float*)d_in[16];
  const float* thW1 = (const float*)d_in[17];
  const float* thb1 = (const float*)d_in[18];
  const float* thW2 = (const float*)d_in[19];
  const float* thb2 = (const float*)d_in[20];
  const float* clW1 = (const float*)d_in[21];
  const float* clb1 = (const float*)d_in[22];
  const float* clW2 = (const float*)d_in[23];
  const float* clb2 = (const float*)d_in[24];
  const float* clW3 = (const float*)d_in[25];
  const float* clb3 = (const float*)d_in[26];

  // ---- workspace carve (256B aligned) ----
  char* ws = (char*)d_ws;
  size_t off = 0;
  auto abf = [&](size_t elems) -> __bf16* {
    void* q = ws + off; off += ((elems * 2 + 255) / 256) * 256; return (__bf16*)q; };
  auto af32 = [&](size_t elems) -> float* {
    void* q = ws + off; off += ((elems * 4 + 255) / 256) * 256; return (float*)q; };

  __bf16* w_ih0 = abf((size_t)G3 * F_FF);
  __bf16* w_hh0 = abf((size_t)G3 * HID);
  __bf16* w_ih1 = abf((size_t)G3 * HID);
  __bf16* w_hh1 = abf((size_t)G3 * HID);
  __bf16* w_dih = abf((size_t)G3 * 32);      // dec_Wih padded K 26->32
  __bf16* w_dhh = abf((size_t)G3 * HID);
  __bf16* w_th1 = abf((size_t)64 * HID);     // (50,512) -> (64,512)
  __bf16* w_th2 = abf((size_t)64 * 64);      // (2,50)   -> (64,64)
  __bf16* w_cl1 = abf((size_t)HID * HID);
  __bf16* w_cl2 = abf((size_t)64 * HID);     // (50,512) -> (64,512)
  __bf16* w_cl3 = abf((size_t)64 * 64);      // (8,50)   -> (64,64)

  __bf16* xa0    = abf((size_t)T_LX * N_BATCH * F_FF);
  __bf16* hzero  = abf((size_t)N_BATCH * HID);
  __bf16* hs0    = abf((size_t)T_LX * N_BATCH * HID);   // layer0 sequence
  __bf16* pp0    = abf((size_t)N_BATCH * HID);
  __bf16* pp1    = abf((size_t)N_BATCH * HID);
  __bf16* decin  = abf((size_t)T_LY * N_BATCH * 32);
  __bf16* hs_dec = abf((size_t)T_LY * N_BATCH * HID);   // decoder sequence
  __bf16* t1buf  = abf((size_t)MROWS * 64);
  __bf16* c1buf  = abf((size_t)MROWS * HID);
  __bf16* c2buf  = abf((size_t)MROWS * 64);
  float*  thbuf  = af32((size_t)MROWS * 2);
  float*  clsbuf = af32((size_t)MROWS * 8);
  unsigned* barw = (unsigned*)af32(64);      // [0]=counter, [1]=generation

  const size_t NH = (size_t)N_BATCH * HID;

  auto cvt = [&](const float* s, __bf16* d, int sr, int sc, int dr, int dc) {
    const int total = dr * dc;
    k_convert_pad<<<(total + 255) / 256, 256, 0, stream>>>(s, d, sr, sc, dr, dc);
  };

  // ---- weight conversion / padding ----
  cvt(Wih0, w_ih0, G3, F_FF, G3, F_FF);
  cvt(Whh0, w_hh0, G3, HID,  G3, HID);
  cvt(Wih1, w_ih1, G3, HID,  G3, HID);
  cvt(Whh1, w_hh1, G3, HID,  G3, HID);
  cvt(dWih, w_dih, G3, 26,   G3, 32);
  cvt(dWhh, w_dhh, G3, HID,  G3, HID);
  cvt(thW1, w_th1, 50, HID,  64, HID);
  cvt(thW2, w_th2, 2,  50,   64, 64);
  cvt(clW1, w_cl1, HID, HID, HID, HID);
  cvt(clW2, w_cl2, 50, HID,  64, HID);
  cvt(clW3, w_cl3, 8,  50,   64, 64);

  // ---- inputs / state init ----
  k_pack_x<<<(T_LX * N_BATCH * F_FF + 255) / 256, 256, 0, stream>>>(x_f, xa0);
  k_build_decin<<<(T_LY * N_BATCH * 32 + 255) / 256, 256, 0, stream>>>(y_t, y_, p_, decin);
  k_zero_bf<<<(int)((NH + 255) / 256), 256, 0, stream>>>(hzero, (int)NH);
  k_init_bar<<<1, 32, 0, stream>>>(&barw[0], &barw[1]);

  const dim3 scanGrid(HID / 64, N_BATCH / 32);  // 64 persistent WGs

  // ---- encoder layer 0: full sequence out ----
  k_gru_scan<<<scanGrid, 256, 0, stream>>>(
      xa0, F_FF, T_LX, /*seq_mode=*/1,
      w_ih0, w_hh0, bih0, bhh0, hzero, hs0, nullptr, nullptr,
      &barw[0], &barw[1]);

  // ---- encoder layer 1: x_t = hs0[t], ping-pong; T even -> final in pp1 ----
  k_gru_scan<<<scanGrid, 256, 0, stream>>>(
      hs0, HID, T_LX, /*seq_mode=*/0,
      w_ih1, w_hh1, bih1, bhh1, hzero, nullptr, pp0, pp1,
      &barw[0], &barw[1]);

  // ---- decoder: 96 steps, full sequence out, h0 = pp1 ----
  k_gru_scan<<<scanGrid, 256, 0, stream>>>(
      decin, 32, T_LY, /*seq_mode=*/1,
      w_dih, w_dhh, dbih, dbhh, pp1, hs_dec, nullptr, nullptr,
      &barw[0], &barw[1]);

  // ---- heads over all LY*N rows ----
  const dim3 g1(1, MROWS / 32), g8(8, MROWS / 32);
  k_gemm<true,  true ><<<g1, 256, 0, stream>>>(hs_dec, HID, w_th1, thb1, 50, t1buf, 64);
  k_gemm<false, false><<<g1, 256, 0, stream>>>(t1buf,  64,  w_th2, thb2, 2,  thbuf, 2);
  k_gemm<true,  true ><<<g8, 256, 0, stream>>>(hs_dec, HID, w_cl1, clb1, HID, c1buf, HID);
  k_gemm<true,  true ><<<g1, 256, 0, stream>>>(c1buf,  HID, w_cl2, clb2, 50,  c2buf, 64);
  k_gemm<false, false><<<g1, 256, 0, stream>>>(c2buf,  64,  w_cl3, clb3, 8,   clsbuf, 8);

  // ---- assemble (N,LY,10) f32 ----
  k_assemble<<<(N_BATCH * T_LY * 10 + 255) / 256, 256, 0, stream>>>(
      thbuf, clsbuf, (float*)d_out);
}
